// SimpleAttention_43430709297645
// MI455X (gfx1250) — matrix operand order repaired and checked by reading, not compiled, not run on hardware
//
#include <hip/hip_runtime.h>

// ---------------- problem constants ----------------
constexpr int kB   = 2;
constexpr int kS   = 1024;
constexpr int kD   = 4096;   // DIM
constexpr int kH   = 32;
constexpr int kHD  = 128;
constexpr int kKVH = 8;
constexpr int kKVD = 1024;
constexpr int kM   = kB * kS;   // 2048 rows

typedef unsigned short ushort_t;
typedef __attribute__((ext_vector_type(16))) __bf16 v16bf;
typedef __attribute__((ext_vector_type(8)))  __bf16 v8bf;
typedef __attribute__((ext_vector_type(8)))  float  v8f;

__device__ __forceinline__ v8f vzero8() {
  v8f z = {0.f,0.f,0.f,0.f,0.f,0.f,0.f,0.f};
  return z;
}

__device__ __forceinline__ ushort_t f2bf(float f) {
  union { float f; unsigned u; } x; x.f = f;
  unsigned r = (x.u + 0x7fffu + ((x.u >> 16) & 1u)) >> 16;
  return (ushort_t)r;
}

__device__ __forceinline__ v8f wmma_bf16(v16bf a, v16bf b, v8f c) {
  // D = A(16x32) * B(32x16) + C, f32 accumulate
  return __builtin_amdgcn_wmma_f32_16x16x32_bf16(false, a, false, b, (short)0, c,
                                                 false, false);
}

// Async global->LDS copy, 16B per lane (GV mode: per-lane 64-bit global addr,
// per-lane 32-bit LDS byte address in the "vdst" operand). Tracked by ASYNCcnt.
__device__ __forceinline__ void async_g2l_b128(unsigned lds_addr, const void* gaddr) {
  asm volatile("global_load_async_to_lds_b128 %0, %1, off"
               :: "v"(lds_addr), "v"((unsigned long long)(size_t)gaddr)
               : "memory");
}
__device__ __forceinline__ void wait_async_le1() {
  asm volatile("s_wait_asynccnt 0x1" ::: "memory");
}
__device__ __forceinline__ void wait_async_le0() {
  asm volatile("s_wait_asynccnt 0x0" ::: "memory");
}

// Build 16-bit A-operand (16x32) for one lane:
//   lane m = lane&15, hi = lane>>4
//   halves j=0..7  -> K = kc + 16*hi + (0..7)
//   halves j=8..15 -> K = kc + 16*hi + 8 + (0..7)
__device__ __forceinline__ v16bf load_a(const ushort_t* rowptr, int kc, int hi) {
  v16bf a;
  ((v8bf*)&a)[0] = *(const v8bf*)(rowptr + kc + 16 * hi);
  ((v8bf*)&a)[1] = *(const v8bf*)(rowptr + kc + 8 + 16 * hi);
  return a;
}

// ---------------- RMSNorm ----------------
__global__ void __launch_bounds__(256)
rmsnorm_k(const float* __restrict__ x, const float* __restrict__ scale,
          float* __restrict__ xn, ushort_t* __restrict__ xnb) {
  const int row = blockIdx.x;
  const int t = threadIdx.x;
  const float* xr = x + (size_t)row * kD;
  float v[16];
  float s = 0.f;
#pragma unroll
  for (int j = 0; j < 16; ++j) {
    v[j] = xr[j * 256 + t];
    s += v[j] * v[j];
  }
#pragma unroll
  for (int off = 16; off; off >>= 1) s += __shfl_xor(s, off, 32);
  __shared__ float red[8];
  const int wid = t >> 5, lane = t & 31;
  if (lane == 0) red[wid] = s;
  __syncthreads();
  float tot = 0.f;
#pragma unroll
  for (int w = 0; w < 8; ++w) tot += red[w];
  const float r = rsqrtf(tot / (float)kD + 1e-8f);
  float* xnr = xn + (size_t)row * kD;
  ushort_t* xbr = xnb + (size_t)row * kD;
#pragma unroll
  for (int j = 0; j < 16; ++j) {
    const int col = j * 256 + t;
    const float o = scale[col] * v[j] * r;
    xnr[col] = o;
    xbr[col] = f2bf(o);
  }
}

// ---------------- fp32 -> bf16 convert ----------------
__global__ void __launch_bounds__(256)
cvt_bf16(const float* __restrict__ in, ushort_t* __restrict__ out, int n) {
  const int i = blockIdx.x * 256 + threadIdx.x;
  if (i < n) out[i] = f2bf(in[i]);
}

// ---------------- RoPE (+ bf16 convert) ----------------
__global__ void __launch_bounds__(256)
rope_bf16(const float* __restrict__ in, ushort_t* __restrict__ out,
          const float* __restrict__ cs, const float* __restrict__ sn,
          int D, int npairs) {
  const int i = blockIdx.x * 256 + threadIdx.x;
  if (i >= npairs) return;
  const int hd = D >> 1;
  const int row = i / hd;
  const int p = i - row * hd;
  const int s = row & (kS - 1);
  const float c = cs[s], si = sn[s];
  const float* pr = in + (size_t)row * D + 2 * p;
  const float t0 = pr[0], t1 = pr[1];
  ushort_t* po = out + (size_t)row * D + 2 * p;
  po[0] = f2bf(t0 * c - t1 * si);
  po[1] = f2bf(t0 * si + t1 * c);
}

// ---------------- V transpose: (b,s,kvh,hd) f32 -> (b,kvh,hd,s) bf16 ----------------
__global__ void __launch_bounds__(256)
vtrans_bf16(const float* __restrict__ in, ushort_t* __restrict__ out, int n) {
  const int i = blockIdx.x * 256 + threadIdx.x;
  if (i >= n) return;
  const int j = i % kKVD;
  const int sidx = (i / kKVD) % kS;
  const int b = i / (kKVD * kS);
  const int kvh = j >> 7;       // /kHD
  const int d = j & (kHD - 1);
  out[((size_t)(b * kKVH + kvh) * kHD + d) * kS + sidx] = f2bf(in[i]);
}

// ---------------- bf16 GEMM: out[M,N] = A[M,K] * W[N,K]^T (+resid) ----------------
// Block-cooperative: block = 8 waves = 128(M) x 64(N) tile. Per 32-K chunk the
// block stages the 64x32 B tile (4KB) into LDS with ONE async b128 per wave
// (ASYNCcnt-tracked, double-buffered) so load(i+1) overlaps compute(i-1..i).
__global__ void __launch_bounds__(256)
gemm_lds(const ushort_t* __restrict__ A, const ushort_t* __restrict__ W,
         float* __restrict__ out, const float* __restrict__ resid,
         int M, int N, int K) {
  __shared__ ushort_t Bs[2][64 * 32];   // 2 x 4KB double buffer
  const int lane = threadIdx.x & 31, wid = threadIdx.x >> 5;
  const int ntn = N >> 6;
  const int mt = blockIdx.x / ntn;
  const int nt = blockIdx.x - mt * ntn;
  const int m0 = mt * 128 + wid * 16;
  const int n0 = nt * 64;
  const int hi = lane >> 4, ln = lane & 15;

  // staging map: 256 lanes x 16B cover 64 rows x 32 kdim (64B per row)
  const int flat = wid * 32 + lane;      // 0..255
  const int sr = flat >> 2;              // row 0..63
  const int sc4 = flat & 3;              // 16B quarter within row
  const ushort_t* gsrc = W + (size_t)(n0 + sr) * K + sc4 * 8;
  const unsigned ldsbase = (unsigned)(size_t)&Bs[0][0];
  const unsigned ldst = ldsbase + (unsigned)(sr * 64 + sc4 * 16);

  v8f c[4];
#pragma unroll
  for (int t = 0; t < 4; ++t) c[t] = vzero8();

  const ushort_t* arow = A + (size_t)(m0 + ln) * K;
  const int nch = K >> 5;

  // prologue: stage chunk 0 into buffer 0
  async_g2l_b128(ldst, gsrc);

  for (int i = 0; i < nch; ++i) {
    const int kc = i << 5;
    __syncthreads();                       // all waves done reading buf[(i-1)&1]
    if (i + 1 < nch) {                     // uniform branch (EXEC stays full)
      async_g2l_b128(ldst + ((i + 1) & 1) * 4096, gsrc + (kc + 32));
      wait_async_le1();                    // in-order => chunk i has landed
    } else {
      wait_async_le0();
    }
    __syncthreads();                       // buf[i&1] visible to every wave

    __builtin_prefetch(arow + kc + 32, 0, 3);   // next A chunk -> global_prefetch_b8
    const v16bf a = load_a(arow, kc, hi);
    const ushort_t* bb = &Bs[i & 1][0];
#pragma unroll
    for (int t = 0; t < 4; ++t) {
      const v16bf bt = *(const v16bf*)(bb + (t * 16 + ln) * 32 + 16 * hi);
      c[t] = wmma_bf16(a, bt, c[t]);
    }
  }

#pragma unroll
  for (int t = 0; t < 4; ++t)
#pragma unroll
    for (int i = 0; i < 8; ++i) {
      const size_t idx = (size_t)(m0 + i + 8 * hi) * N + (n0 + t * 16 + ln);
      float v = c[t][i];
      if (resid) v += resid[idx];
      out[idx] = v;
    }
}

// ---------------- flash attention (one wave = 16 query rows of one head) ----------------
__global__ void __launch_bounds__(256)
attn_fa(const ushort_t* __restrict__ Qb, const ushort_t* __restrict__ Kb,
        const ushort_t* __restrict__ Vt, ushort_t* __restrict__ Ob) {
  __shared__ ushort_t Pb[8][16 * 32];   // per-wave P staging (C-layout -> A-layout)
  const int lane = threadIdx.x & 31, wid = threadIdx.x >> 5;
  const int gw = blockIdx.x * 8 + wid;           // [0, 4096)
  const int qt = gw & 63;
  const int h  = (gw >> 6) & 31;
  const int b  = gw >> 11;
  const int qbase = qt * 16;
  const int kvh = h >> 2;                        // NREP = 4
  const int hi = lane >> 4, ln = lane & 15;
  ushort_t* P = Pb[wid];

  // Q tile: 16 x 128 as 4 A-operands
  const ushort_t* qrow = Qb + (size_t)(b * kS + qbase + ln) * kD + h * kHD;
  v16bf aq[4];
#pragma unroll
  for (int kk = 0; kk < 4; ++kk) aq[kk] = load_a(qrow, kk * 32, hi);

  const ushort_t* kbase = Kb + (size_t)b * kS * kKVD + kvh * kHD;
  const ushort_t* vbase = Vt + (size_t)(b * kKVH + kvh) * kHD * kS;

  const float kNEG = -1e30f;
  float mrow[8], lrow[8];
  v8f o[8];
#pragma unroll
  for (int i = 0; i < 8; ++i) { mrow[i] = kNEG; lrow[i] = 0.f; }
#pragma unroll
  for (int t = 0; t < 8; ++t) o[t] = vzero8();

  const int nch = (qbase + 15) / 32 + 1;
  const float sc = 0.088388347648318447f;       // 1/sqrt(128)

  for (int cix = 0; cix < nch; ++cix) {
    const int kc = cix * 32;
    // ---- S = Q * K^T for 32 keys (two 16-key C tiles) ----
    v8f s0 = vzero8(), s1 = vzero8();
    const ushort_t* k0 = kbase + (size_t)(kc + ln) * kKVD + 16 * hi;
    const ushort_t* k1 = kbase + (size_t)(kc + 16 + ln) * kKVD + 16 * hi;
    if (cix + 1 < nch) {                       // uniform within the wave
      __builtin_prefetch(k0 + (size_t)32 * kKVD, 0, 3);
      __builtin_prefetch(k1 + (size_t)32 * kKVD, 0, 3);
      __builtin_prefetch(vbase + (size_t)ln * kS + kc + 32, 0, 3);
    }
#pragma unroll
    for (int kk = 0; kk < 4; ++kk) {
      const v16bf b0 = *(const v16bf*)(k0 + kk * 32);
      const v16bf b1 = *(const v16bf*)(k1 + kk * 32);
      s0 = wmma_bf16(aq[kk], b0, s0);
      s1 = wmma_bf16(aq[kk], b1, s1);
    }
    // ---- online softmax; stats align with C-layout rows (row = i + 8*hi) ----
#pragma unroll
    for (int i = 0; i < 8; ++i) {
      const int row = qbase + i + 8 * hi;
      const int key0 = kc + ln, key1 = kc + 16 + ln;
      const float f0 = (key0 <= row) ? s0[i] * sc : kNEG;
      const float f1 = (key1 <= row) ? s1[i] * sc : kNEG;
      float mx = fmaxf(f0, f1);
#pragma unroll
      for (int off = 8; off; off >>= 1) mx = fmaxf(mx, __shfl_xor(mx, off, 16));
      const float mn = fmaxf(mrow[i], mx);
      const float corr = __expf(mrow[i] - mn);
      mrow[i] = mn;
      const float p0 = (key0 <= row) ? __expf(f0 - mn) : 0.f;
      const float p1 = (key1 <= row) ? __expf(f1 - mn) : 0.f;
      float rs = p0 + p1;
#pragma unroll
      for (int off = 8; off; off >>= 1) rs += __shfl_xor(rs, off, 16);
      lrow[i] = lrow[i] * corr + rs;
#pragma unroll
      for (int t = 0; t < 8; ++t) o[t][i] *= corr;
      P[(i + 8 * hi) * 32 + ln] = f2bf(p0);
      P[(i + 8 * hi) * 32 + 16 + ln] = f2bf(p1);
    }
    // ---- O += P * V (A from LDS, B from transposed V: contiguous) ----
    v16bf ap;
    ((v8bf*)&ap)[0] = *(const v8bf*)(P + ln * 32 + 16 * hi);
    ((v8bf*)&ap)[1] = *(const v8bf*)(P + ln * 32 + 8 + 16 * hi);
#pragma unroll
    for (int t = 0; t < 8; ++t) {
      const v16bf bv = *(const v16bf*)(vbase + (size_t)(t * 16 + ln) * kS + kc + 16 * hi);
      o[t] = wmma_bf16(ap, bv, o[t]);
    }
  }

  // ---- normalize and store bf16 (b, s, h*HD + d) ----
  ushort_t* orow = Ob + (size_t)(b * kS + qbase) * kD + h * kHD;
#pragma unroll
  for (int t = 0; t < 8; ++t)
#pragma unroll
    for (int i = 0; i < 8; ++i) {
      const float v = o[t][i] / lrow[i];
      orow[(size_t)(i + 8 * hi) * kD + t * 16 + ln] = f2bf(v);
    }
}

// ---------------- host launch ----------------
extern "C" void kernel_launch(void* const* d_in, const int* in_sizes, int n_in,
                              void* d_out, int out_size, void* d_ws, size_t ws_size,
                              hipStream_t stream) {
  const float* x     = (const float*)d_in[0];
  const float* scale = (const float*)d_in[1];
  const float* wq    = (const float*)d_in[2];
  const float* wk    = (const float*)d_in[3];
  const float* wv    = (const float*)d_in[4];
  const float* wo    = (const float*)d_in[5];
  const float* fcos  = (const float*)d_in[6];
  const float* fsin  = (const float*)d_in[7];
  float* out = (float*)d_out;

  char* w = (char*)d_ws;
  size_t off = 0;
  auto alloc = [&](size_t bytes) -> size_t {
    size_t o = off;
    off += (bytes + 255) & ~(size_t)255;
    return o;
  };
  float*    xn   = (float*)   (w + alloc((size_t)kM * kD * 4));
  ushort_t* xnb  = (ushort_t*)(w + alloc((size_t)kM * kD * 2));
  ushort_t* wqb  = (ushort_t*)(w + alloc((size_t)kD * kD * 2));
  ushort_t* wkb  = (ushort_t*)(w + alloc((size_t)kKVD * kD * 2));
  ushort_t* wvb  = (ushort_t*)(w + alloc((size_t)kKVD * kD * 2));
  ushort_t* wob  = (ushort_t*)(w + alloc((size_t)kD * kD * 2));
  float*    qf   = (float*)   (w + alloc((size_t)kM * kD * 4));
  float*    kf   = (float*)   (w + alloc((size_t)kM * kKVD * 4));
  float*    vf   = (float*)   (w + alloc((size_t)kM * kKVD * 4));
  ushort_t* qbf  = (ushort_t*)(w + alloc((size_t)kM * kD * 2));
  ushort_t* kbf  = (ushort_t*)(w + alloc((size_t)kM * kKVD * 2));
  ushort_t* vtb  = (ushort_t*)(w + alloc((size_t)kM * kKVD * 2));
  ushort_t* attnb = xnb;   // xnb dead after projections; reuse for attention output

  // 1) RMSNorm
  rmsnorm_k<<<kM, 256, 0, stream>>>(x, scale, xn, xnb);

  // 2) weights fp32 -> bf16
  const int nqq = kD * kD, nkk = kKVD * kD;
  cvt_bf16<<<(nqq + 255) / 256, 256, 0, stream>>>(wq, wqb, nqq);
  cvt_bf16<<<(nkk + 255) / 256, 256, 0, stream>>>(wk, wkb, nkk);
  cvt_bf16<<<(nkk + 255) / 256, 256, 0, stream>>>(wv, wvb, nkk);
  cvt_bf16<<<(nqq + 255) / 256, 256, 0, stream>>>(wo, wob, nqq);

  // 3) projections (WMMA + async-LDS B staging): block = 128x64 tile
  gemm_lds<<<(kM / 128) * (kD / 64), 256, 0, stream>>>(xnb, wqb, qf, nullptr, kM, kD, kD);
  gemm_lds<<<(kM / 128) * (kKVD / 64), 256, 0, stream>>>(xnb, wkb, kf, nullptr, kM, kKVD, kD);
  gemm_lds<<<(kM / 128) * (kKVD / 64), 256, 0, stream>>>(xnb, wvb, vf, nullptr, kM, kKVD, kD);

  // 4) RoPE + bf16; V transpose
  const int qpairs = kM * (kD / 2), kpairs = kM * (kKVD / 2);
  rope_bf16<<<(qpairs + 255) / 256, 256, 0, stream>>>(qf, qbf, fcos, fsin, kD, qpairs);
  rope_bf16<<<(kpairs + 255) / 256, 256, 0, stream>>>(kf, kbf, fcos, fsin, kKVD, kpairs);
  const int nv = kM * kKVD;
  vtrans_bf16<<<(nv + 255) / 256, 256, 0, stream>>>(vf, vtb, nv);

  // 5) flash attention: waves = B*H*(S/16) = 4096 -> 512 blocks
  attn_fa<<<kB * kH * (kS / 16) / 8, 256, 0, stream>>>(qbf, kbf, vtb, attnb);

  // 6) output projection + residual (adds xn, writes fp32 d_out)
  gemm_lds<<<(kM / 128) * (kD / 64), 256, 0, stream>>>(attnb, wob, out, xn, kM, kD, kD);
}
